// C3PartialConnection_22299470200943
// MI455X (gfx1250) — matrix operand order repaired and checked by reading, not compiled, not run on hardware
//
#include <hip/hip_runtime.h>
#include <math.h>

typedef __attribute__((ext_vector_type(16))) __bf16 v16bf;
typedef __attribute__((ext_vector_type(8)))  float  v8f;

#define H_IN   142
#define W_IN   142
#define H_OUT  138
#define W_OUT  138
#define CIN    6
#define COUT   16
#define ROWS   6                     // output rows per block (138 = 6*23, no waste)
#define INROWS 10                    // ROWS + 4 halo rows
#define KPAD   256                   // 32 (c,dy) groups (30 real) * 8 dx slots
#define CSTRIDE 144                  // LDS row stride (142 cols + pad)
#define ZBASE  (CIN*INROWS*CSTRIDE)  // 8640: ones region base (bias tap + padded K)
#define SIN2   (ZBASE + 896)         // covers worst-case padded-K overrun reads

// group membership bitmasks for the 16 branch outputs (6 triples, 9 quads, 1 sextet)
__device__ __constant__ unsigned char c3_mask[16] = {
    7, 14, 28, 56, 49, 35,              // triples
    15, 30, 60, 57, 51, 39, 27, 54, 45, // quads
    63                                   // sextet
};

__device__ __forceinline__ unsigned short f2bf(float f) {
    unsigned int u = __builtin_bit_cast(unsigned int, f);
    unsigned int r = u + 0x7FFFu + ((u >> 16) & 1u);   // round-to-nearest-even
    return (unsigned short)(r >> 16);
}
__device__ __forceinline__ __bf16 bc16(unsigned short v) {
    return __builtin_bit_cast(__bf16, v);
}
__device__ __forceinline__ float fast_tanh(float z) {
#if __has_builtin(__builtin_amdgcn_tanhf)
    return __builtin_amdgcn_tanhf(z);       // native V_TANH_F32 (CDNA5 TRANS op)
#elif __has_builtin(__builtin_amdgcn_tanh_f32)
    return __builtin_amdgcn_tanh_f32(z);
#else
    return tanhf(z);
#endif
}

// ---------------------------------------------------------------------------
// Pack: dense zero-masked bf16 weight panel W[16][256] (K = group*8 + dx,
// group = c*5+dy, dx padded to 8), all scaled by S = 2/3; slot k=240 carries
// S*bias (multiplied by the 1.0-filled pad region in A). Also the per-half
// group row-base byte-offset table tab[2][16] (tab[h][j] -> group gi = 2j+h).
// ---------------------------------------------------------------------------
__global__ void c3_pack_kernel(const float* __restrict__ w3, const float* __restrict__ b3,
                               const float* __restrict__ w4, const float* __restrict__ b4,
                               const float* __restrict__ w6, const float* __restrict__ b6,
                               unsigned short* __restrict__ wpack,
                               short* __restrict__ tabp) {
    const float S = 2.0f / 3.0f;
    const int tid = threadIdx.x;
    for (int i = tid; i < COUT * KPAD; i += blockDim.x) {
        int o = i >> 8, k = i & 255;
        float val = 0.0f;
        if (k < 240) {                       // 30 real (c,dy) groups
            int g = k >> 3, dx = k & 7;
            if (dx < 5) {
                int c = g / 5, dy = g % 5;
                unsigned mask = c3_mask[o];
                if ((mask >> c) & 1u) {
                    int j = __popc(mask & ((1u << c) - 1u));
                    int rem = dy * 5 + dx;
                    if (o < 6)       val = w3[(o * 3 + j) * 25 + rem];
                    else if (o < 15) val = w4[((o - 6) * 4 + j) * 25 + rem];
                    else             val = w6[j * 25 + rem];
                }
            }
        } else if (k == 240) {               // bias tap (A reads 1.0 there)
            val = (o < 6) ? b3[o] : (o < 15) ? b4[o - 6] : b6[0];
        }
        wpack[i] = f2bf(val * S);
    }
    if (tid < 32) {
        int h = tid >> 4, j = tid & 15;
        int gi = 2 * j + h;
        int byteoff;
        if (gi < 30) {
            int c = gi / 5, dy = gi % 5;
            byteoff = (c * INROWS + dy) * CSTRIDE * 2;   // + (yy*144 + px)*2 at runtime
        } else {
            byteoff = ZBASE * 2;                          // padded groups read the ones region
        }
        tabp[tid] = (short)byteoff;
    }
}

// ---------------------------------------------------------------------------
// Implicit-GEMM conv via bf16 WMMA.
// Block: 320 threads = 10 wave32s; one block per (image, 6-output-row tile).
// Wave w covers output pixels x = 16w..16w+15, all 16 output channels,
// looping over the 6 rows with register-resident B fragments.
// ---------------------------------------------------------------------------
__global__ __launch_bounds__(320)
void c3_conv_wmma_kernel(const float* __restrict__ x,
                         const unsigned short* __restrict__ wpack,
                         const short* __restrict__ tabp,
                         float* __restrict__ out) {
    __shared__ unsigned short sIn[SIN2];       // bf16 tile: 6 ch x 10 rows x 142 cols (+ones)
    __shared__ unsigned short sW[COUT * KPAD]; // bf16 weight panel (scaled, with bias tap)
    __shared__ unsigned short sTab[32];

    const int tid = threadIdx.x;
    const int y0  = blockIdx.x * ROWS;  // first output row of this tile
    const int b   = blockIdx.y;         // image 0..255

    // Stage input rows y0..y0+9 (row-clamped at image edge; clamped rows only
    // feed out-of-range outputs) of all 6 channels, fp32 -> bf16, pairwise.
    // No per-iteration division: p/cr0 fixed per thread, cr strides by 4.
    {
        const float2* xrow = (const float2*)(x + (size_t)b * (CIN * H_IN * W_IN));
        const int PAIRS = W_IN / 2;                          // 71
        const int p   = tid % 80;
        const int cr0 = tid / 80;                            // 0..3
        if (p < PAIRS) {
            for (int cr = cr0; cr < CIN * INROWS; cr += 4) {
                int c = cr / INROWS, r = cr % INROWS;
                int srow = y0 + r; if (srow > H_IN - 1) srow = H_IN - 1;
                float2 v = xrow[((size_t)c * H_IN + srow) * PAIRS + p];
                unsigned int packed = (unsigned int)f2bf(v.x) |
                                      ((unsigned int)f2bf(v.y) << 16);
                *(unsigned int*)&sIn[cr * CSTRIDE + 2 * p] = packed;
            }
        }
    }
    for (int i = tid; i < (SIN2 - ZBASE) / 2; i += 320)
        *(unsigned int*)&sIn[ZBASE + 2 * i] = 0x3F803F80u;   // bf16 1.0 pair
    {   // weight panel copy, 32-bit granularity
        const unsigned int* wp32 = (const unsigned int*)wpack;
        unsigned int* sw32 = (unsigned int*)sW;
        for (int i = tid; i < COUT * KPAD / 2; i += 320) sw32[i] = wp32[i];
    }
    if (tid < 32) sTab[tid] = (unsigned short)tabp[tid];
    __syncthreads();

    const int wv   = tid >> 5;
    const int lane = tid & 31;
    const int half = lane >> 4;     // which K-half this lane holds (A/B layouts)
    const int m    = lane & 15;     // A row (pixel) / B column (channel)
    const int px   = wv * 16 + m;   // this lane's output x (A row)
    if (wv * 16 >= W_OUT) return;   // wave 9 fully out of range (uniform exit)

    // Group row-base byte offsets for this half (gi = 2j + half), into registers.
    int tabv[16];
    #pragma unroll
    for (int j = 0; j < 16; ++j) tabv[j] = (int)sTab[half * 16 + j];

    // B fragments for all 8 K-steps. 16-bit B layout: lane = N,
    // lanes 0-15 hold k_local 0..7,16..23, lanes 16-31 hold 8..15,24..31 (matches A).
    v16bf bfrag[8];
    #pragma unroll
    for (int t = 0; t < 8; ++t) {
        #pragma unroll
        for (int e = 0; e < 16; ++e)
            bfrag[t][e] = bc16(sW[m * KPAD + 32 * t + 8 * half + (e & 7) + 16 * (e >> 3)]);
    }

    #pragma unroll
    for (int yy = 0; yy < ROWS; ++yy) {
        const int y = y0 + yy;
        if (y >= H_OUT) break;          // uniform across block (never fires: 138 = 6*23)

        // Per-lane base pointer: folds in row-in-tile and pixel offsets.
        const char* base = (const char*)sIn + 2 * (yy * CSTRIDE + px);

        v8f acc = {};
        #pragma unroll
        for (int t = 0; t < 8; ++t) {
            // A fragment: two contiguous 8-element bf16 runs
            //   e=0..7  -> group gi = 4t+half   (tabv[2t])
            //   e=8..15 -> group gi = 4t+half+2 (tabv[2t+1])
            union { v16bf v; uint4 q[2]; } au;
            __builtin_memcpy(&au.q[0], base + tabv[2 * t],     16);
            __builtin_memcpy(&au.q[1], base + tabv[2 * t + 1], 16);
            acc = __builtin_amdgcn_wmma_f32_16x16x32_bf16(
                /*neg_a=*/false, au.v, /*neg_b=*/false, bfrag[t],
                /*c_mod=*/(short)0, acc, /*reuse_a=*/false, /*reuse_b=*/false);
        }

        // acc already holds S*(conv + bias); apply amplitude * tanh and store.
        // D layout: VGPR v, lane L -> M = v + 8*(L>=16), N = L&15.
        const int xb0 = wv * 16 + 8 * half;
        float* orow = out + (((size_t)b * COUT + m) * H_OUT + y) * W_OUT;
        float rv[8];
        #pragma unroll
        for (int v = 0; v < 8; ++v)
            rv[v] = 1.7159f * fast_tanh(acc[v]);
        if (xb0 + 8 <= W_OUT) {
            __builtin_memcpy(&orow[xb0], rv, 32);   // 8 consecutive floats
        } else {
            #pragma unroll
            for (int v = 0; v < 8; ++v) {
                int xx = xb0 + v;
                if (xx < W_OUT) orow[xx] = rv[v];
            }
        }
    }
}

// ---------------------------------------------------------------------------
extern "C" void kernel_launch(void* const* d_in, const int* in_sizes, int n_in,
                              void* d_out, int out_size, void* d_ws, size_t ws_size,
                              hipStream_t stream) {
    const float* x  = (const float*)d_in[0];
    const float* w3 = (const float*)d_in[1];
    const float* b3 = (const float*)d_in[2];
    const float* w4 = (const float*)d_in[3];
    const float* b4 = (const float*)d_in[4];
    const float* w6 = (const float*)d_in[5];
    const float* b6 = (const float*)d_in[6];
    float* out = (float*)d_out;

    unsigned short* wpack = (unsigned short*)d_ws;               // 16*256*2 = 8192 B
    short*          tabp  = (short*)((char*)d_ws + 8192);        // 64 B

    c3_pack_kernel<<<1, 256, 0, stream>>>(w3, b3, w4, b4, w6, b6, wpack, tabp);

    dim3 grid(H_OUT / ROWS, 256);   // (row tile, image) = (23, 256)
    c3_conv_wmma_kernel<<<grid, 320, 0, stream>>>(x, wpack, tabp, out);
}
